// GatedAttention_81303730913311
// MI455X (gfx1250) — matrix operand order repaired and checked by reading, not compiled, not run on hardware
//
#include <hip/hip_runtime.h>

// ---------------------------------------------------------------------------
// Fused gated attention for MI455X (gfx1250, wave32, WMMA):
//   S = Doc · Q^T ; U = S · Q ; out = U * Doc      (per batch)
// B=16, Q_LEN=512, DOC_LEN=2048, D=512, fp32 in/out, bf16 WMMA internally.
// One workgroup = 8 waves handles a 64-row Doc block; q tiled by 32 so that
// LDS = 132 KB -> TWO workgroups per 320 KB WGP (4 waves/SIMD) to overlap
// staging with WMMA. LDS regions ordered + base offsets laundered so every
// fragment access is one base VGPR + a 16-bit ds offset immediate.
// ---------------------------------------------------------------------------

typedef __attribute__((ext_vector_type(16))) __bf16 v16bf;
typedef __attribute__((ext_vector_type(8)))  float  v8f;

constexpr int QL = 512;    // Q_LEN
constexpr int DL = 2048;   // DOC_LEN
constexpr int DD = 512;    // D
constexpr int MB = 64;     // doc rows per workgroup
constexpr int QB = 32;     // q rows per iteration

// LDS layout (bytes): hot small regions below the 64 KB ds-offset boundary.
constexpr unsigned Q_OFF   = 0;                      // [32][512] bf16 = 32 KB
constexpr unsigned S_OFF   = Q_OFF  + QB * DD * 2;   // [64][32]  bf16 =  4 KB
constexpr unsigned QT_OFF  = S_OFF  + MB * QB * 2;   // [512][32] bf16 = 32 KB
constexpr unsigned DOC_OFF = QT_OFF + DD * QB * 2;   // [64][512] bf16 = 64 KB
constexpr unsigned SMEM_BYTES = DOC_OFF + MB * DD * 2;  // 135168 -> 2 WGs/WGP

// Keep a region's byte offset opaque so the backend materializes one base
// VGPR (hoisted) and folds only the small constant deltas into ds offsets.
__device__ __forceinline__ unsigned opaque(unsigned off) {
  asm("" : "+v"(off));
  return off;
}

// Hardware bf16 converts (backend emits v_cvt_pk_bf16_f32).
__device__ __forceinline__ unsigned pack_bf16x2(float a, float b) {
  union { __bf16 h[2]; unsigned u; } p;
  p.h[0] = (__bf16)a;
  p.h[1] = (__bf16)b;
  return p.u;
}
__device__ __forceinline__ float bf2f(unsigned short h) {
  return __uint_as_float(((unsigned)h) << 16);
}

// Load a 16-element bf16 fragment as two 16-byte LDS chunks.
__device__ __forceinline__ v16bf frag2(const char* p0, const char* p1) {
  union { v16bf v; uint4 u[2]; } f;
  f.u[0] = *(const uint4*)p0;
  f.u[1] = *(const uint4*)p1;
  return f.v;
}

__device__ __forceinline__ v8f wmma_bf16(v16bf a, v16bf b, v8f c) {
  return __builtin_amdgcn_wmma_f32_16x16x32_bf16(
      /*neg_a=*/false, a, /*neg_b=*/false, b,
      /*c_mod=*/(short)0, c, /*reuse_a=*/false, /*reuse_b=*/false);
}

__global__ __launch_bounds__(256)
void fused_gated_attention(const float* __restrict__ qg,   // (B, 512, 512)
                           const float* __restrict__ dg,   // (B, 2048, 512)
                           float* __restrict__ out) {      // (B, 2048, 512)
  extern __shared__ char smem[];

  const int tid  = threadIdx.x;
  const int lane = tid & 31;
  const int wave = tid >> 5;            // 0..7
  const int wrow = wave & 3;            // wave's 16-row slice of the 64 M rows
  const int wcol = wave >> 2;           // wave's 256-col slice of D
  const int half = lane >> 4;           // 0/1 : K-half per ISA WMMA layouts
  const int ln   = lane & 15;           // row (A) / col (B,C,D) within tile

  const int b  = blockIdx.y;
  const int m0 = blockIdx.x * MB;

  const float* docg  = dg + ((size_t)b * DL + m0) * DD;
  const float* qbase = qg + (size_t)b * QL * DD;

  // Per-thread staging bases (deltas below are i*2048 etc. -> ds immediates).
  char* doc_st = smem + opaque(DOC_OFF + tid * 8);
  char* q_st   = smem + opaque(Q_OFF + tid * 8);
  // qt scatter: d and (qr & 1) are fixed per thread; qr advances by 2 per i.
  const int d0  = (tid * 4) & 511;
  const int qr0 = (tid * 4) >> 9;                 // 0 or 1
  char* qt_st  = smem + opaque(QT_OFF + (d0 * QB + qr0) * 2);

  // ---- stage Doc block (fp32 -> bf16) into LDS, once -----------------------
  #pragma unroll 4
  for (int i = 0; i < 32; ++i) {
    float4 v = *(const float4*)(docg + (size_t)tid * 4 + (size_t)i * 1024);
    uint2 p;
    p.x = pack_bf16x2(v.x, v.y);
    p.y = pack_bf16x2(v.z, v.w);
    *(uint2*)(doc_st + i * 2048) = p;
  }

  v8f acc[16];
  #pragma unroll
  for (int t = 0; t < 16; ++t)
    #pragma unroll
    for (int e = 0; e < 8; ++e) acc[t][e] = 0.0f;

  // Per-lane fragment base pointers (lane row and K-half folded in).
  const int smt = wave >> 1, sqt = wave & 1;        // stage-1 tile of this wave
  const char* s1_abase =
      smem + opaque(DOC_OFF + ((smt * 16 + ln) * DD) * 2 + half * 16);
  const char* s1_bbase =
      smem + opaque(Q_OFF + ((sqt * 16 + ln) * DD) * 2 + half * 16);
  char* s1_sbase = smem + opaque(S_OFF + ((smt * 16 + 8 * half) * QB +
                                          sqt * 16 + ln) * 2);
  const char* s2_abase =
      smem + opaque(S_OFF + ((wrow * 16 + ln) * QB) * 2 + half * 16);
  const char* s2_bbase =
      smem + opaque(QT_OFF + ((wcol * 256 + ln) * QB) * 2 + half * 16);

  for (int qb = 0; qb < QL / QB; ++qb) {
    __syncthreads();   // previous iteration's stage-2 done reading q/qt/s

    // ---- stage Q block into LDS: row-major + transposed (both bf16) --------
    const float* qgb = qbase + (size_t)qb * QB * DD;
    #pragma unroll 4
    for (int i = 0; i < 16; ++i) {
      float4 v = *(const float4*)(qgb + (size_t)tid * 4 + (size_t)i * 1024);
      __bf16 h0 = (__bf16)v.x, h1 = (__bf16)v.y,
             h2 = (__bf16)v.z, h3 = (__bf16)v.w;
      union { __bf16 h[2]; unsigned u; } lo, hi;
      lo.h[0] = h0; lo.h[1] = h1;
      hi.h[0] = h2; hi.h[1] = h3;
      *(uint2*)(q_st + i * 2048) = make_uint2(lo.u, hi.u);
      __bf16* qt = (__bf16*)(qt_st + i * 4);      // qr advances by 2 per i
      qt[0 * QB] = h0;
      qt[1 * QB] = h1;
      qt[2 * QB] = h2;
      qt[3 * QB] = h3;
    }
    // prefetch next Q block into GL2 while we compute
    if (qb + 1 < QL / QB)
      __builtin_prefetch(qgb + (size_t)QB * DD + tid * 64, 0, 1);
    __syncthreads();

    // ---- stage 1: S[0:64,0:32] = Doc_blk · Q_blk^T  (1 tile per wave) ------
    {
      v8f s;
      #pragma unroll
      for (int e = 0; e < 8; ++e) s[e] = 0.0f;
      #pragma unroll 8
      for (int kd = 0; kd < DD / 32; ++kd) {
        v16bf a  = frag2(s1_abase + kd * 64, s1_abase + kd * 64 + 32);
        v16bf bf = frag2(s1_bbase + kd * 64, s1_bbase + kd * 64 + 32);
        s = wmma_bf16(a, bf, s);
      }
      // write S tile to LDS as bf16 (C/D layout: vgpr r -> M = r + 8*half)
      #pragma unroll
      for (int r = 0; r < 8; ++r)
        *(__bf16*)(s1_sbase + r * (QB * 2)) = (__bf16)s[r];
    }
    __syncthreads();

    // ---- stage 2: U[wave slice] += S · Q_blk (K = 32 = one WMMA) -----------
    {
      v16bf a = frag2(s2_abase, s2_abase + 32);
      #pragma unroll
      for (int nt = 0; nt < 16; ++nt) {
        const char* bp = s2_bbase + nt * (16 * QB * 2);   // +nt*1024 bytes
        v16bf bf = frag2(bp, bp + 32);
        acc[nt] = wmma_bf16(a, bf, acc[nt]);
      }
    }
  }

  // ---- epilogue: gate with Doc and store fp32 ------------------------------
  const char* gbase = smem + opaque(DOC_OFF + ((wrow * 16 + 8 * half) * DD +
                                               wcol * 256 + ln) * 2);
  float* obase = out + ((size_t)b * DL + m0 + wrow * 16 + 8 * half) * DD +
                 wcol * 256 + ln;
  #pragma unroll
  for (int nt = 0; nt < 16; ++nt) {
    #pragma unroll
    for (int r = 0; r < 8; ++r) {
      const float gate =
          bf2f(*(const unsigned short*)(gbase + (r * DD + nt * 16) * 2));
      obase[(size_t)r * DD + nt * 16] = acc[nt][r] * gate;
    }
  }
}

extern "C" void kernel_launch(void* const* d_in, const int* in_sizes, int n_in,
                              void* d_out, int out_size, void* d_ws, size_t ws_size,
                              hipStream_t stream) {
  const float* qg = (const float*)d_in[0];   // question_matrix (16, 512, 512)
  const float* dg = (const float*)d_in[1];   // document_matrix (16, 2048, 512)
  float* out = (float*)d_out;                // (16, 2048, 512)
  dim3 grid(DL / MB, 16);                    // 32 doc-blocks x 16 batches
  fused_gated_attention<<<grid, 256, SMEM_BYTES, stream>>>(qg, dg, out);
}
